// SparseMoE_47201690583043
// MI455X (gfx1250) — matrix operand order repaired and checked by reading, not compiled, not run on hardware
//
#include <hip/hip_runtime.h>
#include <hip/hip_bf16.h>
#include <stdint.h>

// Problem constants (from reference): B=8, T=2048, D=1024, E=8, K=2, CAP=4096
#define N_TOK 16384
#define DM    1024
#define DF    4096
#define NEXP  8
#define CAPX  4096

typedef __attribute__((ext_vector_type(16))) __bf16 v16bf;
typedef __attribute__((ext_vector_type(8)))  float  v8f;

union ABFrag { v16bf v; uint4 q[2]; };

// fp32 -> bf16, round-to-nearest-even (inputs are finite)
__device__ __forceinline__ unsigned int f2bf(float f) {
    union { float f; unsigned u; } c; c.f = f;
    unsigned u = c.u + 0x7fffu + ((c.u >> 16) & 1u);
    return u >> 16;
}

__device__ __forceinline__ uint4 pack8_bf16(float4 a, float4 b) {
    uint4 r;
    r.x = f2bf(a.x) | (f2bf(a.y) << 16);
    r.y = f2bf(a.z) | (f2bf(a.w) << 16);
    r.z = f2bf(b.x) | (f2bf(b.y) << 16);
    r.w = f2bf(b.z) | (f2bf(b.w) << 16);
    return r;
}

// ---------------------------------------------------------------------------
// Zero-init of the fp32 output (atomic-add destination)
// ---------------------------------------------------------------------------
__global__ __launch_bounds__(256) void zero_f4(float4* __restrict__ p, int n4) {
    int i = blockIdx.x * 256 + threadIdx.x;
    if (i < n4) p[i] = make_float4(0.f, 0.f, 0.f, 0.f);
}

// ---------------------------------------------------------------------------
// Router: one wave32 per token. logits = x@Wr+br ; noisy = logits + noise*softplus(x@Wn+bn)
// top-2 (lowest index wins ties, matching jax.lax.top_k) + 2-way softmax gates.
// ---------------------------------------------------------------------------
__global__ __launch_bounds__(256) void moe_router(
    const float* __restrict__ x, const float* __restrict__ noise,
    const float* __restrict__ Wr, const float* __restrict__ br,
    const float* __restrict__ Wn, const float* __restrict__ bn,
    int* __restrict__ topi, float* __restrict__ gates)
{
    const int wid  = threadIdx.x >> 5;
    const int lane = threadIdx.x & 31;
    const int t    = blockIdx.x * 8 + wid;

    float aR[8] = {0,0,0,0,0,0,0,0};
    float aN[8] = {0,0,0,0,0,0,0,0};
    const float* xr = x + (size_t)t * DM;
    for (int d = lane; d < DM; d += 32) {
        float xv = xr[d];
        float4 r0 = *(const float4*)(Wr + d * NEXP);
        float4 r1 = *(const float4*)(Wr + d * NEXP + 4);
        float4 n0 = *(const float4*)(Wn + d * NEXP);
        float4 n1 = *(const float4*)(Wn + d * NEXP + 4);
        aR[0] = fmaf(xv, r0.x, aR[0]); aR[1] = fmaf(xv, r0.y, aR[1]);
        aR[2] = fmaf(xv, r0.z, aR[2]); aR[3] = fmaf(xv, r0.w, aR[3]);
        aR[4] = fmaf(xv, r1.x, aR[4]); aR[5] = fmaf(xv, r1.y, aR[5]);
        aR[6] = fmaf(xv, r1.z, aR[6]); aR[7] = fmaf(xv, r1.w, aR[7]);
        aN[0] = fmaf(xv, n0.x, aN[0]); aN[1] = fmaf(xv, n0.y, aN[1]);
        aN[2] = fmaf(xv, n0.z, aN[2]); aN[3] = fmaf(xv, n0.w, aN[3]);
        aN[4] = fmaf(xv, n1.x, aN[4]); aN[5] = fmaf(xv, n1.y, aN[5]);
        aN[6] = fmaf(xv, n1.z, aN[6]); aN[7] = fmaf(xv, n1.w, aN[7]);
    }
    #pragma unroll
    for (int j = 0; j < 8; ++j) {
        float vR = aR[j], vN = aN[j];
        #pragma unroll
        for (int s = 16; s > 0; s >>= 1) {
            vR += __shfl_xor(vR, s, 32);
            vN += __shfl_xor(vN, s, 32);
        }
        aR[j] = vR; aN[j] = vN;
    }
    if (lane == 0) {
        float noisy[8];
        #pragma unroll
        for (int j = 0; j < 8; ++j) {
            float z  = aN[j] + bn[j];
            float sp = (z > 20.0f) ? z : log1pf(expf(z));
            noisy[j] = aR[j] + br[j] + noise[t * NEXP + j] * sp;
        }
        int i0 = 0; float v0 = noisy[0];
        #pragma unroll
        for (int j = 1; j < 8; ++j) if (noisy[j] > v0) { v0 = noisy[j]; i0 = j; }
        int i1 = (i0 == 0) ? 1 : 0; float v1 = noisy[i1];
        #pragma unroll
        for (int j = 0; j < 8; ++j) if (j != i0 && noisy[j] > v1) { v1 = noisy[j]; i1 = j; }
        float ex = expf(v1 - v0);           // <= 1, no overflow
        float g0 = 1.0f / (1.0f + ex);
        topi[t * 2]     = i0;  topi[t * 2 + 1]  = i1;
        gates[t * 2]    = g0;  gates[t * 2 + 1] = 1.0f - g0;
    }
}

// ---------------------------------------------------------------------------
// Per-expert stable compaction in token order (arrival-order rank) with
// capacity drop — exactly the reference semantics. One wave per expert.
// ---------------------------------------------------------------------------
__global__ __launch_bounds__(32) void moe_route_build(
    const int* __restrict__ topi, const float* __restrict__ gates,
    int* __restrict__ idx, float* __restrict__ egate, int* __restrict__ cnt)
{
    const int e    = blockIdx.x;
    const int lane = threadIdx.x;
    int count = 0;
    const unsigned lt = (1u << lane) - 1u;
    for (int t0 = 0; t0 < N_TOK; t0 += 32) {
        int t = t0 + lane;
        int i0 = topi[t * 2], i1 = topi[t * 2 + 1];
        bool f0 = (i0 == e);
        bool flag = f0 || (i1 == e);
        unsigned b = (unsigned)__ballot(flag);
        int rank = count + __popc(b & lt);
        if (flag && rank < CAPX) {
            idx[e * CAPX + rank]   = t;
            egate[e * CAPX + rank] = f0 ? gates[t * 2] : gates[t * 2 + 1];
        }
        count += __popc(b);
    }
    int total = count < CAPX ? count : CAPX;
    for (int r = total + lane; r < CAPX; r += 32) {   // pad: token 0, gate 0
        idx[e * CAPX + r] = 0;
        egate[e * CAPX + r] = 0.f;
    }
    if (lane == 0) cnt[e] = total;
}

// ---------------------------------------------------------------------------
// Layer 1: h = relu(gather(x)[128x1024] @ W1e[1024x4096] + b1e), bf16 out.
// 256 thr = 8 waves; block tile 128x128, wave tile 64x32 (4x2 WMMA 16x16x32 bf16).
// ---------------------------------------------------------------------------
__global__ __launch_bounds__(256) void expert_ffn1(
    const float* __restrict__ x,
    const float* __restrict__ W1e,
    const float* __restrict__ b1e,
    const int*   __restrict__ idxe,
    const int*   __restrict__ cnte,
    unsigned short* __restrict__ h)
{
    __shared__ unsigned short sA[128][40];    // 128 x 32 bf16, padded stride
    __shared__ unsigned short sB[32][136];    // 32  x 128 bf16 (K-major), padded
    __shared__ unsigned short sS[128][136];   // bf16 transpose stage for coalesced h stores

    const int cnt   = *cnte;
    const int mbase = blockIdx.y * 128;
    if (mbase >= cnt) return;                 // tile fully beyond capacity-used rows
    const int nbase = blockIdx.x * 128;

    const int tid   = threadIdx.x;
    const int lane  = tid & 31;
    const int wid   = tid >> 5;
    const int waveM = (wid >> 2) * 64;        // 0 / 64
    const int waveN = (wid & 3) * 32;         // 0 / 32 / 64 / 96
    const int half  = lane >> 4;
    const int l16   = lane & 15;

    // A gather: thread -> (row, 16-wide K half)
    const int arow = tid >> 1;
    const int akh  = (tid & 1) * 16;
    const int tok  = idxe[mbase + arow];      // padded rows -> token 0 (finite)
    const float* xr = x + (size_t)tok * DM + akh;

    // B: thread -> (K row, 16-wide N chunk)
    const int brow = tid >> 3;
    const int bcol = (tid & 7) * 16;
    const float* wp = W1e + (size_t)brow * DF + nbase + bcol;

    v8f acc[4][2] = {};

    #pragma unroll 1
    for (int kk = 0; kk < DM; kk += 32) {
        // speculative prefetch of next K-slab (weights dominate HBM traffic)
        __builtin_prefetch(wp + (size_t)(kk + 32) * DF, 0, 1);
        __builtin_prefetch(xr + kk + 32, 0, 1);
        __syncthreads();
        {   // A tile: 16 fp32 -> 16 bf16 as two b128 LDS stores
            const float4* p = (const float4*)(xr + kk);
            float4 f0 = p[0], f1 = p[1], f2 = p[2], f3 = p[3];
            *(uint4*)&sA[arow][akh]     = pack8_bf16(f0, f1);
            *(uint4*)&sA[arow][akh + 8] = pack8_bf16(f2, f3);
        }
        {   // B tile
            const float4* p = (const float4*)(wp + (size_t)kk * DF);
            float4 g0 = p[0], g1 = p[1], g2 = p[2], g3 = p[3];
            *(uint4*)&sB[brow][bcol]     = pack8_bf16(g0, g1);
            *(uint4*)&sB[brow][bcol + 8] = pack8_bf16(g2, g3);
        }
        __syncthreads();

        ABFrag af[4], bfr[2];
        #pragma unroll
        for (int i = 0; i < 4; ++i) {   // A frag: lane<16 M-rows, K halves per lane-half
            const unsigned short* pa = &sA[waveM + i * 16 + l16][half * 8];
            af[i].q[0] = *(const uint4*)pa;          // K = half*8 .. +7
            af[i].q[1] = *(const uint4*)(pa + 16);   // K = 16+half*8 .. +7
        }
        #pragma unroll
        for (int j = 0; j < 2; ++j) {   // B frag: lane<16 N-cols, K halves per lane-half (same K order as A)
            const unsigned short* pb = &sB[half * 8][waveN + j * 16 + l16];
            #pragma unroll
            for (int s = 0; s < 16; s += 2) {
                const int k = (s < 8) ? s : s + 8;
                ((unsigned int*)&bfr[j])[s >> 1] =
                    (unsigned int)pb[k * 136] | ((unsigned int)pb[(k + 1) * 136] << 16);
            }
        }
        #pragma unroll
        for (int i = 0; i < 4; ++i)
            #pragma unroll
            for (int j = 0; j < 2; ++j)
                acc[i][j] = __builtin_amdgcn_wmma_f32_16x16x32_bf16(
                    false, af[i].v, false, bfr[j].v, (short)0, acc[i][j], false, false);
    }

    // bias + relu, stage as bf16 in [m][n] layout
    #pragma unroll
    for (int j = 0; j < 2; ++j) {
        const float bj = b1e[nbase + waveN + j * 16 + l16];
        #pragma unroll
        for (int i = 0; i < 4; ++i) {
            #pragma unroll
            for (int r = 0; r < 8; ++r) {
                float v = acc[i][j][r] + bj;
                v = v > 0.0f ? v : 0.0f;
                sS[waveM + i * 16 + half * 8 + r][waveN + j * 16 + l16] =
                    (unsigned short)f2bf(v);
            }
        }
    }
    __syncthreads();
    {   // coalesced bf16 store of h: each thread 64 contiguous elements
        const int r = tid >> 1;
        const int c = (tid & 1) * 64;
        unsigned short* hp = h + (size_t)(mbase + r) * DF + nbase + c;
        const unsigned short* sp = &sS[r][c];
        #pragma unroll
        for (int q = 0; q < 8; ++q)
            *(uint4*)(hp + q * 8) = *(const uint4*)(sp + q * 8);
    }
}

// ---------------------------------------------------------------------------
// Layer 2: out_tok += gate * (h[128x4096] @ W2e[4096x1024] + b2e), fp32 atomics.
// h tile (already bf16) is moved with CDNA5 async global->LDS copies (ASYNCcnt),
// overlapping the copy with the W2 fp32->bf16 conversion path.
// ---------------------------------------------------------------------------
__global__ __launch_bounds__(256) void expert_ffn2(
    const unsigned short* __restrict__ h,
    const float* __restrict__ W2e,
    const float* __restrict__ b2e,
    const int*   __restrict__ idxe,
    const float* __restrict__ gatee,
    const int*   __restrict__ cnte,
    float* __restrict__ out)
{
    __shared__ unsigned short sA[128][40];
    __shared__ unsigned short sB[32][136];

    const int cnt   = *cnte;
    const int mbase = blockIdx.y * 128;
    if (mbase >= cnt) return;
    const int nbase = blockIdx.x * 128;

    const int tid   = threadIdx.x;
    const int lane  = tid & 31;
    const int wid   = tid >> 5;
    const int waveM = (wid >> 2) * 64;
    const int waveN = (wid & 3) * 32;
    const int half  = lane >> 4;
    const int l16   = lane & 15;

    const int arow = tid >> 1;
    const int akh  = (tid & 1) * 16;
    const unsigned short* hr = h + (size_t)(mbase + arow) * DF + akh;
    // LDS byte offset of this thread's A-tile slot (generic shared address:
    // high 32 bits = shared aperture, low 32 bits = wave LDS offset)
    const unsigned ldsA = (unsigned)(uintptr_t)&sA[arow][akh];

    const int brow = tid >> 3;
    const int bcol = (tid & 7) * 16;
    const float* wp = W2e + (size_t)brow * DM + nbase + bcol;

    v8f acc[4][2] = {};

    #pragma unroll 1
    for (int kk = 0; kk < DF; kk += 32) {
        __builtin_prefetch(wp + (size_t)(kk + 32) * DM, 0, 1);
        __syncthreads();
        {   // A tile: async global->LDS, 32B/lane as two b128 ops.
            // INST_OFFSET is added to BOTH the LDS and global addresses (ISA 10.x),
            // so offset:16 covers the second 16B without extra address math.
            const unsigned long long gp = (unsigned long long)(uintptr_t)(hr + kk);
            asm volatile("global_load_async_to_lds_b128 %0, %1, off"
                         :: "v"(ldsA), "v"(gp) : "memory");
            asm volatile("global_load_async_to_lds_b128 %0, %1, off offset:16"
                         :: "v"(ldsA), "v"(gp) : "memory");
        }
        {   // B tile: convert W2 fp32 -> bf16 (overlaps the async copy above)
            const float4* p = (const float4*)(wp + (size_t)kk * DM);
            float4 g0 = p[0], g1 = p[1], g2 = p[2], g3 = p[3];
            *(uint4*)&sB[brow][bcol]     = pack8_bf16(g0, g1);
            *(uint4*)&sB[brow][bcol + 8] = pack8_bf16(g2, g3);
        }
        asm volatile("s_wait_asynccnt 0" ::: "memory");
        __syncthreads();

        ABFrag af[4], bfr[2];
        #pragma unroll
        for (int i = 0; i < 4; ++i) {
            const unsigned short* pa = &sA[waveM + i * 16 + l16][half * 8];
            af[i].q[0] = *(const uint4*)pa;
            af[i].q[1] = *(const uint4*)(pa + 16);
        }
        #pragma unroll
        for (int j = 0; j < 2; ++j) {
            const unsigned short* pb = &sB[half * 8][waveN + j * 16 + l16];
            #pragma unroll
            for (int s = 0; s < 16; s += 2) {
                const int k = (s < 8) ? s : s + 8;
                ((unsigned int*)&bfr[j])[s >> 1] =
                    (unsigned int)pb[k * 136] | ((unsigned int)pb[(k + 1) * 136] << 16);
            }
        }
        #pragma unroll
        for (int i = 0; i < 4; ++i)
            #pragma unroll
            for (int j = 0; j < 2; ++j)
                acc[i][j] = __builtin_amdgcn_wmma_f32_16x16x32_bf16(
                    false, af[i].v, false, bfr[j].v, (short)0, acc[i][j], false, false);
    }

    // gated scatter-add; padded/dropped rows have gate == 0 and are skipped
    const float bj0 = b2e[nbase + waveN + l16];
    const float bj1 = b2e[nbase + waveN + 16 + l16];
    #pragma unroll
    for (int i = 0; i < 4; ++i) {
        #pragma unroll
        for (int r = 0; r < 8; ++r) {
            const int m = waveM + i * 16 + half * 8 + r;
            const float g = gatee[mbase + m];
            if (g != 0.0f) {
                float* op = out + (size_t)idxe[mbase + m] * DM + nbase + waveN + l16;
                atomicAdd(op,      (acc[i][0][r] + bj0) * g);
                atomicAdd(op + 16, (acc[i][1][r] + bj1) * g);
            }
        }
    }
}

// ---------------------------------------------------------------------------
extern "C" void kernel_launch(void* const* d_in, const int* in_sizes, int n_in,
                              void* d_out, int out_size, void* d_ws, size_t ws_size,
                              hipStream_t stream) {
    (void)in_sizes; (void)n_in; (void)out_size; (void)ws_size;
    const float* x     = (const float*)d_in[0];
    const float* noise = (const float*)d_in[1];
    const float* Wr    = (const float*)d_in[2];
    const float* br    = (const float*)d_in[3];
    const float* Wn    = (const float*)d_in[4];
    const float* bn    = (const float*)d_in[5];
    const float* W1    = (const float*)d_in[6];
    const float* b1    = (const float*)d_in[7];
    const float* W2    = (const float*)d_in[8];
    const float* b2    = (const float*)d_in[9];
    float* out = (float*)d_out;

    // Workspace carve-out (~34.1 MB total)
    char* w = (char*)d_ws;
    int*   topi  = (int*)w;    w += (size_t)N_TOK * 2 * sizeof(int);
    float* gates = (float*)w;  w += (size_t)N_TOK * 2 * sizeof(float);
    int*   idx   = (int*)w;    w += (size_t)NEXP * CAPX * sizeof(int);
    float* egate = (float*)w;  w += (size_t)NEXP * CAPX * sizeof(float);
    int*   cnt   = (int*)w;    w += 256;
    unsigned short* h = (unsigned short*)w;   // CAPX * DF bf16 = 33.5 MB, reused per expert

    const int out4 = N_TOK * DM / 4;
    zero_f4<<<(out4 + 255) / 256, 256, 0, stream>>>((float4*)out, out4);
    moe_router<<<N_TOK / 8, 256, 0, stream>>>(x, noise, Wr, br, Wn, bn, topi, gates);
    moe_route_build<<<NEXP, 32, 0, stream>>>(topi, gates, idx, egate, cnt);

    // Expert-serialized (stream-ordered) so h stays hot in the 192MB L2 between
    // ffn1 and ffn2 of each expert; weights per expert (33.5MB) also L2-resident.
    for (int e = 0; e < NEXP; ++e) {
        expert_ffn1<<<dim3(DF / 128, CAPX / 128), 256, 0, stream>>>(
            x, W1 + (size_t)e * DM * DF, b1 + (size_t)e * DF,
            idx + e * CAPX, cnt + e, h);
        expert_ffn2<<<dim3(DM / 128, CAPX / 128), 256, 0, stream>>>(
            h, W2 + (size_t)e * DF * DM, b2 + (size_t)e * DM,
            idx + e * CAPX, egate + e * CAPX, cnt + e, out);
    }
}
